// SAGENet_57157424775339
// MI455X (gfx1250) — matrix operand order repaired and checked by reading, not compiled, not run on hardware
//
#include <hip/hip_runtime.h>

// ---------------------------------------------------------------------------
// CDNA5 (gfx1250) GraphSAGE autoencoder pipeline.
// Precision plan: activations bf16 (WMMA inputs), aggregation + epilogue f32.
// GEMMs use v_wmma_f32_16x16x32_bf16 with 32x64 register tiles per wave
// (8 WMMAs per 12 b128 fragment loads); scatter uses native f32 atomics
// on an L2-resident (102 MB < 192 MB) accumulation buffer.
// ---------------------------------------------------------------------------

typedef __attribute__((ext_vector_type(16))) __bf16 v16bf;
typedef __attribute__((ext_vector_type(8)))  __bf16 v8bf;
typedef __attribute__((ext_vector_type(8)))  float  v8f;

#define DEV __device__ __forceinline__

// ---------------- fragment loaders (per ISA 7.12.2 layouts) ----------------

// A matrix 16x32 bf16: lane&15 = row M; lanes 0-15 hold K = k0..k0+7 (v0..v3)
// and K = k0+16..k0+23 (v4..v7); lanes 16-31 hold the +8 K offsets.
DEV v16bf load_frag_a(const __bf16* __restrict__ A, int K, int row_base,
                      int kk, int lane, int M) {
    int m = row_base + (lane & 15);
    if (m >= M) m = M - 1;                       // edge clamp (store is guarded)
    int k0 = kk + ((lane >> 4) << 3);            // +0 or +8
    const __bf16* p = A + (size_t)m * K + k0;
    union { v16bf f; v8bf h[2]; } u;
    u.h[0] = *(const v8bf*)(p);                  // K = k0   .. k0+7
    u.h[1] = *(const v8bf*)(p + 16);             // K = k0+16.. k0+23
    return u.f;
}

// B matrix 32x16 bf16: lane&15 = column N; lanes 0-15 hold K=kk..kk+7 and
// kk+16..kk+23, lanes 16-31 hold the +8 K offsets (same order as A).
// Bt is stored transposed [N][K].
DEV v16bf load_frag_b(const __bf16* __restrict__ Bt, int K, int col_base,
                      int kk, int lane) {
    int n = col_base + (lane & 15);
    int k0 = kk + ((lane >> 4) << 3);            // +0 or +8
    const __bf16* p = Bt + (size_t)n * K + k0;
    union { v16bf f; v8bf h[2]; } u;
    u.h[0] = *(const v8bf*)(p);
    u.h[1] = *(const v8bf*)(p + 16);
    return u.f;
}

// ---------------- fused dual-GEMM:  C = A(M x K) * Bt^T + bias -------------
// block = 128 threads (4 waves) arranged 2x2; block tile = 64 rows x 128 cols.
// Each wave owns a 32x64 register tile: 2 A-frags x 4 B-frags -> 8 WMMAs/K-step.
__global__ void gemm_wmma_kernel(const __bf16* __restrict__ A,
                                 const __bf16* __restrict__ Bt,
                                 const float*  __restrict__ bias,
                                 __bf16* __restrict__ Cb,
                                 float*  __restrict__ Cf,
                                 int M, int N, int K, int do_relu, int out_f32) {
    const int lane = threadIdx.x & 31;
    const int wave = threadIdx.x >> 5;
    const int row0 = blockIdx.x * 64 + (wave & 1) * 32;   // 2 row tiles of 16
    const int col0 = blockIdx.y * 128 + (wave >> 1) * 64; // 4 col tiles of 16
    if (row0 >= M) return;                                 // wave-uniform exit

    v8f acc[2][4] = {};

    for (int kk = 0; kk < K; kk += 32) {
        // batch all 12 fragment loads so they clause & overlap the WMMAs
        v16bf a0 = load_frag_a(A, K, row0,      kk, lane, M);
        v16bf a1 = load_frag_a(A, K, row0 + 16, kk, lane, M);
        v16bf b0 = load_frag_b(Bt, K, col0,      kk, lane);
        v16bf b1 = load_frag_b(Bt, K, col0 + 16, kk, lane);
        v16bf b2 = load_frag_b(Bt, K, col0 + 32, kk, lane);
        v16bf b3 = load_frag_b(Bt, K, col0 + 48, kk, lane);

        acc[0][0] = __builtin_amdgcn_wmma_f32_16x16x32_bf16(
                        false, a0, false, b0, (short)0, acc[0][0], false, false);
        acc[0][1] = __builtin_amdgcn_wmma_f32_16x16x32_bf16(
                        false, a0, false, b1, (short)0, acc[0][1], false, false);
        acc[0][2] = __builtin_amdgcn_wmma_f32_16x16x32_bf16(
                        false, a0, false, b2, (short)0, acc[0][2], false, false);
        acc[0][3] = __builtin_amdgcn_wmma_f32_16x16x32_bf16(
                        false, a0, false, b3, (short)0, acc[0][3], false, false);
        acc[1][0] = __builtin_amdgcn_wmma_f32_16x16x32_bf16(
                        false, a1, false, b0, (short)0, acc[1][0], false, false);
        acc[1][1] = __builtin_amdgcn_wmma_f32_16x16x32_bf16(
                        false, a1, false, b1, (short)0, acc[1][1], false, false);
        acc[1][2] = __builtin_amdgcn_wmma_f32_16x16x32_bf16(
                        false, a1, false, b2, (short)0, acc[1][2], false, false);
        acc[1][3] = __builtin_amdgcn_wmma_f32_16x16x32_bf16(
                        false, a1, false, b3, (short)0, acc[1][3], false, false);
    }

    // C/D layout: VGPR i -> row (i + (lane<16 ? 0 : 8)), col = lane&15
    const int half = (lane >> 4) << 3;          // 0 or 8
    const int ncol = lane & 15;
#pragma unroll
    for (int r = 0; r < 2; ++r) {
        const int mbase = row0 + r * 16 + half;
#pragma unroll
        for (int c = 0; c < 4; ++c) {
            const int n = col0 + c * 16 + ncol;
            const float bi = bias[n];
#pragma unroll
            for (int i = 0; i < 8; ++i) {
                int m = mbase + i;
                if (m >= M) continue;
                float v = acc[r][c][i] + bi;
                if (do_relu) v = fmaxf(v, 0.0f);
                if (out_f32) Cf[(size_t)m * N + n] = v;
                else         Cb[(size_t)m * N + n] = (__bf16)v;
            }
        }
    }
}

// ---------------- helper kernels -------------------------------------------

__global__ void zero_f32_kernel(float* __restrict__ p, size_t n) {
    size_t t = (size_t)blockIdx.x * blockDim.x + threadIdx.x;
    if (t < n) p[t] = 0.0f;
}

__global__ void init_h_kernel(const float* __restrict__ attr,
                              __bf16* __restrict__ h, size_t n) {
    size_t t = (size_t)blockIdx.x * blockDim.x + threadIdx.x;
    if (t < n) h[t] = (__bf16)attr[t];
}

// overwrite masked rows with mask token (block per masked node, 128 threads)
__global__ void mask_kernel(const int* __restrict__ mask_nodes,
                            const float* __restrict__ token,
                            __bf16* __restrict__ h, int d) {
    int node = mask_nodes[blockIdx.x];
    h[(size_t)node * d + threadIdx.x] = (__bf16)token[threadIdx.x];
}

__global__ void deg_kernel(const int* __restrict__ dst,
                           float* __restrict__ deg, int E) {
    int e = blockIdx.x * blockDim.x + threadIdx.x;
    if (e < E) unsafeAtomicAdd(&deg[dst[e]], 1.0f);
}

// build transposed+concatenated weights: wt[n][k] = (k<din? Ws[k][n] : Wn[k-din][n])
__global__ void prep_w_kernel(const float* __restrict__ ws,
                              const float* __restrict__ wn,
                              __bf16* __restrict__ wt, int din, int dout) {
    size_t t = (size_t)blockIdx.x * blockDim.x + threadIdx.x;
    size_t total = (size_t)dout * 2 * din;
    if (t >= total) return;
    int twoK = 2 * din;
    int n = (int)(t / twoK);
    int k = (int)(t % twoK);
    float v = (k < din) ? ws[(size_t)k * dout + n]
                        : wn[(size_t)(k - din) * dout + n];
    wt[t] = (__bf16)v;
}

// scatter-add: agg[dst[e]][j..j+3] += h[src[e]][j..j+3]  (f32 atomics, L2-resident)
__global__ void scatter_kernel(const __bf16* __restrict__ h,
                               const int* __restrict__ src,
                               const int* __restrict__ dst,
                               float* __restrict__ agg,
                               long long total, int log_d4, int d) {
    long long t = (long long)blockIdx.x * blockDim.x + threadIdx.x;
    if (t >= total) return;
    int e = (int)(t >> log_d4);
    int j = (int)((t & ((1 << log_d4) - 1)) << 2);
    int s = src[e];
    int dd = dst[e];
    const __bf16* hp = h + (size_t)s * d + j;
    float* ap = agg + (size_t)dd * d + j;
    unsafeAtomicAdd(ap + 0, (float)hp[0]);
    unsafeAtomicAdd(ap + 1, (float)hp[1]);
    unsafeAtomicAdd(ap + 2, (float)hp[2]);
    unsafeAtomicAdd(ap + 3, (float)hp[3]);
}

// hcat[i][0:din] = h[i][:]; hcat[i][din:2din] = bf16(agg[i][:] / max(deg,1))
__global__ void hcat_kernel(const __bf16* __restrict__ h,
                            const float* __restrict__ agg,
                            const float* __restrict__ deg,
                            __bf16* __restrict__ hcat,
                            size_t total, int log_din, int din) {
    size_t t = (size_t)blockIdx.x * blockDim.x + threadIdx.x;
    if (t >= total) return;
    int i = (int)(t >> log_din);
    int j = (int)(t & (din - 1));
    float dinv = 1.0f / fmaxf(deg[i], 1.0f);
    size_t base = (size_t)i * (2 * din);
    hcat[base + j]       = h[(size_t)i * din + j];
    hcat[base + din + j] = (__bf16)(agg[(size_t)i * din + j] * dinv);
}

// masked MSE: block per masked node, 128 threads (= IN_DIM)
__global__ void mse_kernel(const float* __restrict__ attr,
                           const float* __restrict__ dec,
                           const int* __restrict__ mask_nodes,
                           float* __restrict__ out, int d, float inv_count) {
    __shared__ float red[128];
    int node = mask_nodes[blockIdx.x];
    size_t idx = (size_t)node * d + threadIdx.x;
    float diff = dec[idx] - attr[idx];
    red[threadIdx.x] = diff * diff * inv_count;
    __syncthreads();
    for (int s = 64; s > 0; s >>= 1) {
        if (threadIdx.x < s) red[threadIdx.x] += red[threadIdx.x + s];
        __syncthreads();
    }
    if (threadIdx.x == 0) unsafeAtomicAdd(out, red[0]);
}

// ---------------- host orchestration ---------------------------------------

static inline int ilog2i(int x) { int l = 0; while ((1 << l) < x) ++l; return l; }

extern "C" void kernel_launch(void* const* d_in, const int* in_sizes, int n_in,
                              void* d_out, int out_size, void* d_ws, size_t ws_size,
                              hipStream_t stream) {
    const int IN_DIM = 128;
    const int N = in_sizes[0] / IN_DIM;
    const int E = in_sizes[2];
    const int NUM_MASK = in_sizes[4];

    const float* attr       = (const float*)d_in[0];
    const float* mask_token = (const float*)d_in[1];
    const int*   src        = (const int*)d_in[2];
    const int*   dst        = (const int*)d_in[3];
    const int*   mask_nodes = (const int*)d_in[4];
    // per-layer weights: w_self, w_neigh, bias
    const float* wS[4] = { (const float*)d_in[5],  (const float*)d_in[8],
                           (const float*)d_in[11], (const float*)d_in[14] };
    const float* wN[4] = { (const float*)d_in[6],  (const float*)d_in[9],
                           (const float*)d_in[12], (const float*)d_in[15] };
    const float* bb[4] = { (const float*)d_in[7],  (const float*)d_in[10],
                           (const float*)d_in[13], (const float*)d_in[16] };
    const int dins[4]  = {128, 512, 512, 512};
    const int douts[4] = {512, 512, 512, 128};

    // workspace carve-out
    size_t off = 0;
    auto carve = [&](size_t bytes) {
        void* p = (char*)d_ws + off;
        off += (bytes + 255) & ~(size_t)255;
        return p;
    };
    __bf16* h    = (__bf16*)carve((size_t)N * 512 * sizeof(__bf16));   // activations
    __bf16* hcat = (__bf16*)carve((size_t)N * 1024 * sizeof(__bf16));  // [h | h_neigh]
    float*  agg  = (float*) carve((size_t)N * 512 * sizeof(float));    // f32 agg / decoded
    __bf16* wt   = (__bf16*)carve((size_t)1024 * 512 * sizeof(__bf16));// W^T concat bf16
    float*  deg  = (float*) carve((size_t)N * sizeof(float));
    (void)ws_size; (void)n_in; (void)out_size;

    const int TB = 256;
    auto blocks = [](long long n, int tb) { return (unsigned)((n + tb - 1) / tb); };

    // h = bf16(attr); h[mask_nodes] = mask_token
    init_h_kernel<<<blocks((long long)N * IN_DIM, TB), TB, 0, stream>>>(
        attr, h, (size_t)N * IN_DIM);
    mask_kernel<<<NUM_MASK, IN_DIM, 0, stream>>>(mask_nodes, mask_token, h, IN_DIM);

    // degree (shared by all layers)
    zero_f32_kernel<<<blocks(N, TB), TB, 0, stream>>>(deg, (size_t)N);
    deg_kernel<<<blocks(E, TB), TB, 0, stream>>>(dst, deg, E);

    for (int L = 0; L < 4; ++L) {
        const int din = dins[L], dout = douts[L], twoK = 2 * din;

        // weights -> concatenated transposed bf16  [dout][2*din]
        prep_w_kernel<<<blocks((long long)dout * twoK, TB), TB, 0, stream>>>(
            wS[L], wN[L], wt, din, dout);

        // agg = 0; agg[dst] += h[src]
        zero_f32_kernel<<<blocks((long long)N * din, TB), TB, 0, stream>>>(
            agg, (size_t)N * din);
        long long sc_total = (long long)E * (din >> 2);
        scatter_kernel<<<blocks(sc_total, TB), TB, 0, stream>>>(
            h, src, dst, agg, sc_total, ilog2i(din >> 2), din);

        // hcat = [h | agg/max(deg,1)]
        hcat_kernel<<<blocks((long long)N * din, TB), TB, 0, stream>>>(
            h, agg, deg, hcat, (size_t)N * din, ilog2i(din), din);

        // fused GEMM: out = hcat (N x 2K) * wt^T + b   [+relu]
        const int relu = (L == 0 || L == 2) ? 1 : 0;
        const int outf32 = (L == 3) ? 1 : 0;          // decoded stays f32
        dim3 grid((N + 63) / 64, dout / 128);
        gemm_wmma_kernel<<<grid, 128, 0, stream>>>(
            hcat, wt, bb[L], h, agg /* decoded reuses agg */,
            N, dout, twoK, relu, outf32);
    }

    // masked MSE -> d_out[0]
    zero_f32_kernel<<<1, 32, 0, stream>>>((float*)d_out, 1);
    const float inv = 1.0f / ((float)NUM_MASK * (float)IN_DIM);
    mse_kernel<<<NUM_MASK, IN_DIM, 0, stream>>>(
        attr, agg, mask_nodes, (float*)d_out, IN_DIM, inv);
}
